// QCircuitAlgorithmModel_90211493085952
// MI455X (gfx1250) — matrix operand order repaired and checked by reading, not compiled, not run on hardware
//
#include <hip/hip_runtime.h>
#include <hip/hip_bf16.h>

typedef __attribute__((ext_vector_type(16))) _Float16 v16h;
typedef __attribute__((ext_vector_type(8)))  _Float16 v8h;
typedef __attribute__((ext_vector_type(8)))  float    v8f;

#define NN 131072
#define EE 2097152
#define BBATCH 256
#define HID 64

__device__ __forceinline__ float silu_f(float x) { return x / (1.0f + __expf(-x)); }

// order-preserving float<->uint encoding so atomicMax(uint) == max(float)
__device__ __forceinline__ unsigned f2ord(float f) {
  unsigned u = __float_as_uint(f);
  return (u & 0x80000000u) ? ~u : (u | 0x80000000u);
}
__device__ __forceinline__ float ord2f(unsigned u) {
  return (u & 0x80000000u) ? __uint_as_float(u & 0x7fffffffu) : __uint_as_float(~u);
}

// ---------------------------------------------------------------------------
// Build node features as f16 [N, 64] (cols 0-15 gate emb, 16-31 qubit emb sum,
// 32 gate_index_norm, 33-63 zero padding for the WMMA K dimension)
// ---------------------------------------------------------------------------
__global__ void build_features_kernel(const int* __restrict__ gti,
                                      const int* __restrict__ qidx,
                                      const unsigned char* __restrict__ isdir,
                                      const int* __restrict__ arity,
                                      const float* __restrict__ gnorm,
                                      const float* __restrict__ gate_emb,
                                      const float* __restrict__ qubit_emb,
                                      const float* __restrict__ pos_emb,
                                      _Float16* __restrict__ h16)
{
  int i = blockIdx.x * blockDim.x + threadIdx.x;
  if (i >= NN) return;
  const float* ge = gate_emb + gti[i] * 16;
  float qp[16];
#pragma unroll
  for (int j = 0; j < 16; ++j) qp[j] = 0.0f;
  int ar = arity[i];
  bool dir = isdir[i] != 0;
#pragma unroll
  for (int a = 0; a < 3; ++a) {
    if (a < ar) {
      const float* qe = qubit_emb + qidx[i * 3 + a] * 16;
      const float* pe = pos_emb + a * 16;
#pragma unroll
      for (int j = 0; j < 16; ++j) qp[j] += qe[j] * (dir ? pe[j] : 0.0625f);
    }
  }
  _Float16 row[64];
#pragma unroll
  for (int j = 0; j < 16; ++j) row[j] = (_Float16)ge[j];
#pragma unroll
  for (int j = 0; j < 16; ++j) row[16 + j] = (_Float16)qp[j];
  row[32] = (_Float16)gnorm[i];
#pragma unroll
  for (int j = 33; j < 64; ++j) row[j] = (_Float16)0.0f;
  v8h* dst = (v8h*)(h16 + (size_t)i * 64);
#pragma unroll
  for (int c = 0; c < 8; ++c) dst[c] = ((const v8h*)row)[c];
}

// ---------------------------------------------------------------------------
// Pack W[din,64] (f32, row-major [K][n]) into the WMMA B-fragment layout:
// out[((nt*2+ks)*32 + lane)*16 + t] = f16( W[ks*32 + (lane>>4)*8 + t + (t>=8 ? 8 : 0)][nt*16 + (lane&15)] )
// K padded with zeros to 64.
// ---------------------------------------------------------------------------
__global__ void pack_w_kernel(const float* __restrict__ W, int din,
                              _Float16* __restrict__ out)
{
  int t = threadIdx.x;          // 0..255
  int lane = t & 31;
  int ks = (t >> 5) & 1;
  int nt = t >> 6;
  int n = (nt << 4) + (lane & 15);
  int kbase = ks * 32 + ((lane >> 4) << 3);
  _Float16* o = out + t * 16;
#pragma unroll
  for (int i = 0; i < 16; ++i) {
    int K = kbase + i + ((i >> 3) << 3);
    o[i] = (_Float16)((K < din) ? W[(size_t)K * 64 + n] : 0.0f);
  }
}

// ---------------------------------------------------------------------------
// Fused projection GEMM: computes ALL FOUR of q/k/v/skip in one pass.
//   O*[N,64] = A[N,64](f16) @ B*(packed f16) + bias*
// Block = 256 threads = 8 waves; wave w: M-tile = blockIdx.x*2 + w/4, N-tile = w%4.
// Each wave loads its A fragment ONCE and drives 4 independent WMMA
// accumulator chains (8 x v_wmma_f32_16x16x32_f16 per wave), letting the
// scheduler hide D->C latency across chains instead of inserting v_nops.
// Grid exact (N/32 blocks) -> EXEC always all-ones for WMMA.
// ---------------------------------------------------------------------------
__global__ void gemm64_qkvs_wmma_kernel(const _Float16* __restrict__ A,
                                        const _Float16* __restrict__ Bq,
                                        const _Float16* __restrict__ Bk,
                                        const _Float16* __restrict__ Bv,
                                        const _Float16* __restrict__ Bs,
                                        const float* __restrict__ bq,
                                        const float* __restrict__ bk,
                                        const float* __restrict__ bv,
                                        const float* __restrict__ bs,
                                        float* __restrict__ Oq,
                                        float* __restrict__ Ok,
                                        float* __restrict__ Ov,
                                        float* __restrict__ Os)
{
  int wave = threadIdx.x >> 5;
  int lane = threadIdx.x & 31;
  int mtile = blockIdx.x * 2 + (wave >> 2);
  int ntile = wave & 3;
  int mbase = mtile << 4;

  // A fragment base: row m = mbase + (lane&15), half-offset (lane>=16 ? 8 : 0)
  const _Float16* arow = A + (size_t)(mbase + (lane & 15)) * 64 + ((lane >> 4) << 3);

  v8f cq = {}, ck = {}, cv = {}, cs = {};
#pragma unroll
  for (int ks = 0; ks < 2; ++ks) {
    v8h a0 = *(const v8h*)(arow + ks * 32);        // K = koff+lo+0..7
    v8h a1 = *(const v8h*)(arow + ks * 32 + 16);   // K = koff+lo+16..23
    v16h a;
#pragma unroll
    for (int i = 0; i < 8; ++i) { a[i] = a0[i]; a[8 + i] = a1[i]; }

    size_t bo = ((size_t)(ntile * 2 + ks) * 32 + lane) * 16;
    v16h b0 = *(const v16h*)(Bq + bo);
    v16h b1 = *(const v16h*)(Bk + bo);
    v16h b2 = *(const v16h*)(Bv + bo);
    v16h b3 = *(const v16h*)(Bs + bo);

    cq = __builtin_amdgcn_wmma_f32_16x16x32_f16(false, a, false, b0, (short)0, cq, false, false);
    ck = __builtin_amdgcn_wmma_f32_16x16x32_f16(false, a, false, b1, (short)0, ck, false, false);
    cv = __builtin_amdgcn_wmma_f32_16x16x32_f16(false, a, false, b2, (short)0, cv, false, false);
    cs = __builtin_amdgcn_wmma_f32_16x16x32_f16(false, a, false, b3, (short)0, cs, false, false);
  }

  int col = (ntile << 4) + (lane & 15);
  int row0 = mbase + ((lane >> 4) << 3);
  float vbq = bq[col], vbk = bk[col], vbv = bv[col], vbs = bs[col];
#pragma unroll
  for (int r = 0; r < 8; ++r) {
    size_t o = (size_t)(row0 + r) * 64 + col;
    Oq[o] = cq[r] + vbq;
    Ok[o] = ck[r] + vbk;
    Ov[o] = cv[r] + vbv;
    Os[o] = cs[r] + vbs;
  }
}

// ---------------------------------------------------------------------------
// Edge pass 1: one wave per edge. score[e] = dot(q[dst], k[src]) / 8;
// atomicMax(ordered encoding) into mx[dst].
// ---------------------------------------------------------------------------
__global__ void edge_score_max_kernel(const int* __restrict__ ei,
                                      const float* __restrict__ q,
                                      const float* __restrict__ k,
                                      float* __restrict__ score,
                                      unsigned* __restrict__ mx)
{
  int e = (blockIdx.x * blockDim.x + threadIdx.x) >> 5;
  int lane = threadIdx.x & 31;
  int s = ei[e];
  int d = ei[EE + e];
  float2 a = ((const float2*)(q + (size_t)d * 64))[lane];
  float2 b = ((const float2*)(k + (size_t)s * 64))[lane];
  float p = a.x * b.x + a.y * b.y;
#pragma unroll
  for (int off = 16; off > 0; off >>= 1) p += __shfl_xor(p, off, 32);
  if (lane == 0) {
    float sc = p * 0.125f;   // 1/sqrt(64)
    score[e] = sc;
    atomicMax(mx + d, f2ord(sc));
  }
}

// ---------------------------------------------------------------------------
// Edge pass 2: one thread per edge. a = exp(score - m[dst]); score[e] = a;
// denom[dst] += a.
// ---------------------------------------------------------------------------
__global__ void edge_exp_denom_kernel(const int* __restrict__ ei,
                                      const unsigned* __restrict__ mx,
                                      float* __restrict__ score,
                                      float* __restrict__ denom)
{
  int e = blockIdx.x * blockDim.x + threadIdx.x;
  if (e >= EE) return;
  int d = ei[EE + e];
  float m = ord2f(mx[d]);
  float a = __expf(score[e] - m);
  score[e] = a;
  atomicAdd(denom + d, a);
}

// ---------------------------------------------------------------------------
// Edge pass 3: one wave per edge. acc[dst] += (a/denom[dst]) * v[src]
// ---------------------------------------------------------------------------
__global__ void edge_scatter_kernel(const int* __restrict__ ei,
                                    const float* __restrict__ score,
                                    const float* __restrict__ denom,
                                    const float* __restrict__ v,
                                    float* __restrict__ acc)
{
  int e = (blockIdx.x * blockDim.x + threadIdx.x) >> 5;
  int lane = threadIdx.x & 31;
  int s = ei[e];
  int d = ei[EE + e];
  float dn = denom[d];
  dn = dn > 0.0f ? dn : 1.0f;
  float w = score[e] / dn;
  float2 vv = ((const float2*)(v + (size_t)s * 64))[lane];
  atomicAdd(acc + (size_t)d * 64 + lane * 2,     w * vv.x);
  atomicAdd(acc + (size_t)d * 64 + lane * 2 + 1, w * vv.y);
}

// ---------------------------------------------------------------------------
// Epilogue: y = silu(acc) in-place (f32), also write f16 copy for next layer
// ---------------------------------------------------------------------------
__global__ void silu_convert_kernel(float* __restrict__ acc,
                                    _Float16* __restrict__ h16)
{
  int i = blockIdx.x * blockDim.x + threadIdx.x;
  if (i >= NN) return;
  float* row = acc + (size_t)i * 64;
  _Float16 hrow[64];
#pragma unroll
  for (int j = 0; j < 64; ++j) {
    float y = silu_f(row[j]);
    row[j] = y;
    hrow[j] = (_Float16)y;
  }
  v8h* dst = (v8h*)(h16 + (size_t)i * 64);
#pragma unroll
  for (int c = 0; c < 8; ++c) dst[c] = ((const v8h*)hrow)[c];
}

// ---------------------------------------------------------------------------
// Global mean pool (accumulation): one wave per node, 2 channels per lane
// ---------------------------------------------------------------------------
__global__ void pool_kernel(const float* __restrict__ x,
                            const int* __restrict__ batch,
                            float* __restrict__ sums,
                            float* __restrict__ cnts)
{
  int t = blockIdx.x * blockDim.x + threadIdx.x;
  int node = t >> 5;
  int lane = t & 31;
  if (node >= NN) return;
  int b = batch[node];
  float2 vv = ((const float2*)(x + (size_t)node * 64))[lane];
  atomicAdd(sums + (size_t)b * 64 + lane * 2,     vv.x);
  atomicAdd(sums + (size_t)b * 64 + lane * 2 + 1, vv.y);
  if (lane == 0) atomicAdd(cnts + b, 1.0f);
}

// ---------------------------------------------------------------------------
// Head stage A: combined[b,0:64] = pooled/cnt ; combined[b,64:128] = silu(gf@gmlp)
// ---------------------------------------------------------------------------
__global__ void head_a_kernel(const float* __restrict__ pooled,
                              const float* __restrict__ cnts,
                              const float* __restrict__ gf,
                              const float* __restrict__ gmlpW,
                              const float* __restrict__ gmlpB,
                              float* __restrict__ combined)
{
  int t = blockIdx.x * blockDim.x + threadIdx.x;
  if (t >= BBATCH * 64) return;
  int b = t >> 6;
  int j = t & 63;
  float cnt = cnts[b];
  cnt = cnt > 1.0f ? cnt : 1.0f;
  combined[(size_t)b * 128 + j] = pooled[(size_t)b * 64 + j] / cnt;
  float s = gmlpB[j];
#pragma unroll 4
  for (int kk = 0; kk < 28; ++kk) s += gf[b * 28 + kk] * gmlpW[kk * 64 + j];
  combined[(size_t)b * 128 + 64 + j] = silu_f(s);
}

__global__ void head_h1_kernel(const float* __restrict__ combined,
                               const float* __restrict__ W1,
                               const float* __restrict__ b1,
                               float* __restrict__ h1)
{
  int t = blockIdx.x * blockDim.x + threadIdx.x;
  if (t >= BBATCH * 128) return;
  int b = t >> 7;
  int j = t & 127;
  float s = b1[j];
  const float* crow = combined + (size_t)b * 128;
#pragma unroll 8
  for (int kk = 0; kk < 128; ++kk) s += crow[kk] * W1[kk * 128 + j];
  h1[(size_t)b * 128 + j] = silu_f(s);
}

__global__ void head_h2_kernel(const float* __restrict__ h1,
                               const float* __restrict__ W2,
                               const float* __restrict__ b2,
                               float* __restrict__ h2)
{
  int t = blockIdx.x * blockDim.x + threadIdx.x;
  if (t >= BBATCH * 64) return;
  int b = t >> 6;
  int j = t & 63;
  float s = b2[j];
  const float* hrow = h1 + (size_t)b * 128;
#pragma unroll 8
  for (int kk = 0; kk < 128; ++kk) s += hrow[kk] * W2[kk * 64 + j];
  h2[(size_t)b * 64 + j] = silu_f(s);
}

__global__ void head_out_kernel(const float* __restrict__ h2,
                                const float* __restrict__ hW,
                                const float* __restrict__ hB,
                                float* __restrict__ out)
{
  int t = blockIdx.x * blockDim.x + threadIdx.x;
  if (t >= BBATCH * 2) return;
  int b = t >> 1;
  int j = t & 1;
  float s = hB[j];
  const float* hrow = h2 + (size_t)b * 64;
#pragma unroll 8
  for (int kk = 0; kk < 64; ++kk) s += hrow[kk] * hW[kk * 2 + j];
  out[(size_t)b * 2 + j] = s;
}

// ---------------------------------------------------------------------------
// Host launcher
// ---------------------------------------------------------------------------
extern "C" void kernel_launch(void* const* d_in, const int* in_sizes, int n_in,
                              void* d_out, int out_size, void* d_ws, size_t ws_size,
                              hipStream_t stream)
{
  (void)in_sizes; (void)n_in; (void)out_size; (void)ws_size;

  const int* gti   = (const int*)d_in[0];
  const int* qidx  = (const int*)d_in[1];
  const unsigned char* isdir = (const unsigned char*)d_in[2];
  const int* arity = (const int*)d_in[3];
  const float* gnorm = (const float*)d_in[4];
  const int* ei    = (const int*)d_in[5];
  const int* batch = (const int*)d_in[6];
  const float* gf  = (const float*)d_in[7];
  const float* gate_emb  = (const float*)d_in[8];
  const float* qubit_emb = (const float*)d_in[9];
  const float* pos_emb   = (const float*)d_in[10];
  const float* Wq0 = (const float*)d_in[11]; const float* bq0 = (const float*)d_in[12];
  const float* Wk0 = (const float*)d_in[13]; const float* bk0 = (const float*)d_in[14];
  const float* Wv0 = (const float*)d_in[15]; const float* bv0 = (const float*)d_in[16];
  const float* Ws0 = (const float*)d_in[17]; const float* bs0 = (const float*)d_in[18];
  const float* Wq1 = (const float*)d_in[19]; const float* bq1 = (const float*)d_in[20];
  const float* Wk1 = (const float*)d_in[21]; const float* bk1 = (const float*)d_in[22];
  const float* Wv1 = (const float*)d_in[23]; const float* bv1 = (const float*)d_in[24];
  const float* Ws1 = (const float*)d_in[25]; const float* bs1 = (const float*)d_in[26];
  const float* gmlpW = (const float*)d_in[27]; const float* gmlpB = (const float*)d_in[28];
  const float* bbW1 = (const float*)d_in[29]; const float* bbB1 = (const float*)d_in[30];
  const float* bbW2 = (const float*)d_in[31]; const float* bbB2 = (const float*)d_in[32];
  const float* headW = (const float*)d_in[33]; const float* headB = (const float*)d_in[34];
  float* out = (float*)d_out;

  // ---- workspace partition ----
  char* p = (char*)d_ws;
  auto walloc = [&](size_t bytes) -> void* {
    void* r = (void*)p;
    p += (bytes + 255) & ~(size_t)255;
    return r;
  };
  _Float16* h16   = (_Float16*)walloc((size_t)NN * 64 * sizeof(_Float16));
  float*    q     = (float*)   walloc((size_t)NN * 64 * sizeof(float));
  float*    k     = (float*)   walloc((size_t)NN * 64 * sizeof(float));
  float*    v     = (float*)   walloc((size_t)NN * 64 * sizeof(float));
  float*    acc   = (float*)   walloc((size_t)NN * 64 * sizeof(float));
  float*    score = (float*)   walloc((size_t)EE * sizeof(float));
  unsigned* mx    = (unsigned*)walloc((size_t)NN * sizeof(unsigned));
  float*    denom = (float*)   walloc((size_t)NN * sizeof(float));
  float*    pooled= (float*)   walloc((size_t)BBATCH * 64 * sizeof(float));
  float*    cnts  = (float*)   walloc((size_t)BBATCH * sizeof(float));
  float*    comb  = (float*)   walloc((size_t)BBATCH * 128 * sizeof(float));
  float*    h1    = (float*)   walloc((size_t)BBATCH * 128 * sizeof(float));
  float*    h2    = (float*)   walloc((size_t)BBATCH * 64 * sizeof(float));
  _Float16* wpk   = (_Float16*)walloc((size_t)8 * 4096 * sizeof(_Float16));

  _Float16* wq0p = wpk + 0 * 4096;
  _Float16* wk0p = wpk + 1 * 4096;
  _Float16* wv0p = wpk + 2 * 4096;
  _Float16* ws0p = wpk + 3 * 4096;
  _Float16* wq1p = wpk + 4 * 4096;
  _Float16* wk1p = wpk + 5 * 4096;
  _Float16* wv1p = wpk + 6 * 4096;
  _Float16* ws1p = wpk + 7 * 4096;

  // ---- pack weights into WMMA B-fragment layout ----
  pack_w_kernel<<<1, 256, 0, stream>>>(Wq0, 33, wq0p);
  pack_w_kernel<<<1, 256, 0, stream>>>(Wk0, 33, wk0p);
  pack_w_kernel<<<1, 256, 0, stream>>>(Wv0, 33, wv0p);
  pack_w_kernel<<<1, 256, 0, stream>>>(Ws0, 33, ws0p);
  pack_w_kernel<<<1, 256, 0, stream>>>(Wq1, 64, wq1p);
  pack_w_kernel<<<1, 256, 0, stream>>>(Wk1, 64, wk1p);
  pack_w_kernel<<<1, 256, 0, stream>>>(Wv1, 64, wv1p);
  pack_w_kernel<<<1, 256, 0, stream>>>(Ws1, 64, ws1p);

  // ---- node features ----
  build_features_kernel<<<NN / 256, 256, 0, stream>>>(
      gti, qidx, isdir, arity, gnorm, gate_emb, qubit_emb, pos_emb, h16);

  const int gemmBlocks  = NN / 32;       // 8 waves/block, 2 Mtiles x 4 Ntiles
  const int edgeWaveBlk = EE / 8;        // wave-per-edge, 8 waves/block
  const int edgeThrBlk  = EE / 256;

  const _Float16* wqp[2] = {wq0p, wq1p};
  const _Float16* wkp[2] = {wk0p, wk1p};
  const _Float16* wvp[2] = {wv0p, wv1p};
  const _Float16* wsp[2] = {ws0p, ws1p};
  const float* bqs[2] = {bq0, bq1};
  const float* bks[2] = {bk0, bk1};
  const float* bvs[2] = {bv0, bv1};
  const float* bss[2] = {bs0, bs1};

  for (int layer = 0; layer < 2; ++layer) {
    // fused q/k/v/skip projections: one A read, 8 WMMAs per wave
    gemm64_qkvs_wmma_kernel<<<gemmBlocks, 256, 0, stream>>>(
        h16, wqp[layer], wkp[layer], wvp[layer], wsp[layer],
        bqs[layer], bks[layer], bvs[layer], bss[layer],
        q, k, v, acc);

    hipMemsetAsync(mx, 0, (size_t)NN * sizeof(unsigned), stream);
    hipMemsetAsync(denom, 0, (size_t)NN * sizeof(float), stream);

    edge_score_max_kernel<<<edgeWaveBlk, 256, 0, stream>>>(ei, q, k, score, mx);
    edge_exp_denom_kernel<<<edgeThrBlk, 256, 0, stream>>>(ei, mx, score, denom);
    edge_scatter_kernel<<<edgeWaveBlk, 256, 0, stream>>>(ei, score, denom, v, acc);

    // silu + convert to f16 for next layer (acc keeps f32 result)
    silu_convert_kernel<<<NN / 256, 256, 0, stream>>>(acc, h16);
  }

  // ---- global mean pool ----
  hipMemsetAsync(pooled, 0, (size_t)BBATCH * 64 * sizeof(float), stream);
  hipMemsetAsync(cnts, 0, (size_t)BBATCH * sizeof(float), stream);
  pool_kernel<<<(NN * 32) / 256, 256, 0, stream>>>(acc, batch, pooled, cnts);

  // ---- heads ----
  head_a_kernel<<<(BBATCH * 64) / 256, 256, 0, stream>>>(pooled, cnts, gf, gmlpW, gmlpB, comb);
  head_h1_kernel<<<(BBATCH * 128) / 256, 256, 0, stream>>>(comb, bbW1, bbB1, h1);
  head_h2_kernel<<<(BBATCH * 64) / 256, 256, 0, stream>>>(h1, bbW2, bbB2, h2);
  head_out_kernel<<<2, 256, 0, stream>>>(h2, headW, headB, out);
}
